// SlotGenerator_49460843381715
// MI455X (gfx1250) — matrix operand order repaired and checked by reading, not compiled, not run on hardware
//
#include <hip/hip_runtime.h>
#include <math.h>

// ---------------------------------------------------------------------------
// Types for CDNA5 WMMA (wave32, v_wmma_f32_16x16x32_bf16)
// ---------------------------------------------------------------------------
typedef __attribute__((ext_vector_type(16))) __bf16  v16bf;
typedef __attribute__((ext_vector_type(8)))  float   v8f;
typedef __attribute__((ext_vector_type(4)))  unsigned int u32x4;

union FragBF { u32x4 u[2]; v16bf v; };

// Native f32->bf16 conversion (fptrunc float->bfloat, RTNE; lowers to
// hardware cvt on gfx1250 instead of manual bit-twiddling).
__device__ __forceinline__ unsigned short f2bf(float f) {
  union { __bf16 h; unsigned short u; } x;
  x.h = (__bf16)f;
  return x.u;
}
__device__ __forceinline__ unsigned pack2bf(float a, float b) {
  union { __bf16 h[2]; unsigned u; } x;
  x.h[0] = (__bf16)a; x.h[1] = (__bf16)b;
  return x.u;
}

// ---------------------------------------------------------------------------
// Generic batched WMMA GEMM:  D[b] = scale * (A[b] x B[b]) (+bias[m]) (+=)
//   A: MxK  (AT: stored [K][M] row-major)
//   B: KxN  (BT: stored [N][K] row-major)
//   D: MxN  (DT: stored transposed, element (m,n) at n*ldd+m)
//   ADDD: accumulate into D; GM: guard rows against M (only when M%32 != 0)
// Invariants relied upon (true for every launch below):
//   K % 32 == 0, N % 64 == 0, all pointers/ld's 16-byte aligned.
// Block: 256 threads = 8 waves, tile 32(M) x 64(N), K-step 32.
// ---------------------------------------------------------------------------
template<bool AT, bool BT, bool DT, bool ADDD, bool GM>
__global__ __launch_bounds__(256) void gemm_wmma_bf16(
    const float* __restrict__ A, long sA, int lda,
    const float* __restrict__ B, long sB, int ldb,
    float* __restrict__ D, long sD, int ldd,
    const float* __restrict__ bias,
    int M, int N, int K, float scale)
{
  __shared__ __align__(16) unsigned short As[32][40];   // [m][k], 80B rows
  __shared__ __align__(16) unsigned short Bs[64][40];   // [n][k] (K-contig)

  const int bz   = blockIdx.z;
  const float* Ab = A + (long)bz * sA;
  const float* Bb = B + (long)bz * sB;
  float*       Db = D + (long)bz * sD;

  const int m0   = blockIdx.y * 32;
  const int n0   = blockIdx.x * 64;
  const int tid  = threadIdx.x;
  const int lane = tid & 31;
  const int wave = tid >> 5;
  const int mt   = wave >> 2;      // 0..1  (wave's 16-row tile)
  const int nt   = wave & 3;       // 0..3  (wave's 16-col tile)
  const int r    = lane & 15;
  const int hi   = lane >> 4;

  v8f acc = {};

  for (int k0 = 0; k0 < K; k0 += 32) {
    // ---- stage A tile 32x32 ----
    if (!AT) {
      // row-major: 4 contiguous K per thread -> 1x b128 load, 1x b64 ds store
      int m  = tid >> 3;
      int kk = (tid & 7) * 4;
      float4 v = make_float4(0.f, 0.f, 0.f, 0.f);
      if (!GM || (m0 + m) < M)
        v = *reinterpret_cast<const float4*>(Ab + (long)(m0 + m) * lda + k0 + kk);
      uint2 p; p.x = pack2bf(v.x, v.y); p.y = pack2bf(v.z, v.w);
      *reinterpret_cast<uint2*>(&As[m][kk]) = p;
    } else {
      // col-major source: 4 contiguous M per thread, scatter into As rows
      int k  = tid >> 3;
      int m4 = (tid & 7) * 4;
      float4 v = *reinterpret_cast<const float4*>(Ab + (long)(k0 + k) * lda + m0 + m4);
      As[m4 + 0][k] = f2bf(v.x);
      As[m4 + 1][k] = f2bf(v.y);
      As[m4 + 2][k] = f2bf(v.z);
      As[m4 + 3][k] = f2bf(v.w);
    }
    // ---- stage B tile 32x64 -> Bs[n][k] ----
    if (BT) {
      // [N][K] source: 8 contiguous K per thread -> 2x b128 load, 1x b128 ds store
      int n  = tid >> 2;
      int k8 = (tid & 3) * 8;
      const float4 v0 = *reinterpret_cast<const float4*>(Bb + (long)(n0 + n) * ldb + k0 + k8);
      const float4 v1 = *reinterpret_cast<const float4*>(Bb + (long)(n0 + n) * ldb + k0 + k8 + 4);
      u32x4 p;
      p.x = pack2bf(v0.x, v0.y); p.y = pack2bf(v0.z, v0.w);
      p.z = pack2bf(v1.x, v1.y); p.w = pack2bf(v1.z, v1.w);
      *reinterpret_cast<u32x4*>(&Bs[n][k8]) = p;
    } else {
      // [K][N] source: 8 contiguous N per thread -> 2x b128 load, transpose scatter
      int k  = tid >> 3;
      int n8 = (tid & 7) * 8;
      const float4 v0 = *reinterpret_cast<const float4*>(Bb + (long)(k0 + k) * ldb + n0 + n8);
      const float4 v1 = *reinterpret_cast<const float4*>(Bb + (long)(k0 + k) * ldb + n0 + n8 + 4);
      Bs[n8 + 0][k] = f2bf(v0.x);
      Bs[n8 + 1][k] = f2bf(v0.y);
      Bs[n8 + 2][k] = f2bf(v0.z);
      Bs[n8 + 3][k] = f2bf(v0.w);
      Bs[n8 + 4][k] = f2bf(v1.x);
      Bs[n8 + 5][k] = f2bf(v1.y);
      Bs[n8 + 6][k] = f2bf(v1.z);
      Bs[n8 + 7][k] = f2bf(v1.w);
    }
    // prefetch next K-slab of B (CDNA5 global_prefetch_b8)
    if (tid == 0 && k0 + 32 < K) {
      const float* p = BT ? &Bb[(long)n0 * ldb + (k0 + 32)]
                          : &Bb[(long)(k0 + 32) * ldb + n0];
      __builtin_prefetch(p, 0, 0);
    }
    __syncthreads();

    // ---- fragments per ISA 16-bit layouts ----
    FragBF fa, fb;
    // A 16x32: lanes 0-15 -> K {0..7,16..23}; lanes 16-31 -> K {8..15,24..31}
    fa.u[0] = *reinterpret_cast<const u32x4*>(&As[mt * 16 + r][hi * 8]);
    fa.u[1] = *reinterpret_cast<const u32x4*>(&As[mt * 16 + r][16 + hi * 8]);
    // B 32x16: lanes 0-15 -> K {0..7,16..23}; lanes 16-31 -> K {8..15,24..31} (col = r)
    fb.u[0] = *reinterpret_cast<const u32x4*>(&Bs[nt * 16 + r][hi * 8]);
    fb.u[1] = *reinterpret_cast<const u32x4*>(&Bs[nt * 16 + r][16 + hi * 8]);

    acc = __builtin_amdgcn_wmma_f32_16x16x32_bf16(
        false, fa.v, false, fb.v, (short)0, acc, false, false);
    __syncthreads();
  }

  // ---- store per C/D layout: VGPR i -> row (hi?8:0)+i, col = r ----
#pragma unroll
  for (int i = 0; i < 8; ++i) {
    int row = m0 + mt * 16 + hi * 8 + i;
    int col = n0 + nt * 16 + r;
    if (!GM || row < M) {
      float v = acc[i] * scale;
      if (bias) v += bias[row];
      float* dst = DT ? (Db + (long)col * ldd + row)
                      : (Db + (long)row * ldd + col);
      if (ADDD) *dst += v; else *dst = v;
    }
  }
}

// ---------------------------------------------------------------------------
// Elementwise / reduction helper kernels
// ---------------------------------------------------------------------------
__global__ void k_zero(float* p, long n) {
  long i = (long)blockIdx.x * blockDim.x + threadIdx.x;
  if (i < n) p[i] = 0.f;
}
__global__ void k_copy(const float* s, float* d, long n) {
  long i = (long)blockIdx.x * blockDim.x + threadIdx.x;
  if (i < n) d[i] = s[i];
}

// softmax over 16 slots (axis=1) per (b, n); logits/masks: [B,16,N]
__global__ void k_softmax16(const float* __restrict__ logits,
                            float* __restrict__ masks, int N, long total) {
  long t = (long)blockIdx.x * blockDim.x + threadIdx.x;
  if (t >= total) return;
  int b = (int)(t / N), n = (int)(t % N);
  const float* p = logits + (long)b * 16 * N + n;
  float m = -1e30f;
#pragma unroll
  for (int l = 0; l < 16; ++l) m = fmaxf(m, p[(long)l * N]);
  float s = 0.f;
#pragma unroll
  for (int l = 0; l < 16; ++l) s += __expf(p[(long)l * N] - m);
  float inv = 1.f / s;
  float* q = masks + (long)b * 16 * N + n;
#pragma unroll
  for (int l = 0; l < 16; ++l) q[(long)l * N] = __expf(p[(long)l * N] - m) * inv;
}

// row softmax: one block (256 thr) per row of length L
__global__ __launch_bounds__(256) void k_row_softmax(
    const float* __restrict__ in, float* __restrict__ out, int L) {
  long row = blockIdx.x;
  const float* p = in + row * (long)L;
  float* q = out + row * (long)L;
  __shared__ float red[256];
  int tid = threadIdx.x;
  float m = -1e30f;
  for (int i = tid; i < L; i += 256) m = fmaxf(m, p[i]);
  red[tid] = m; __syncthreads();
  for (int s = 128; s > 0; s >>= 1) {
    if (tid < s) red[tid] = fmaxf(red[tid], red[tid + s]);
    __syncthreads();
  }
  m = red[0]; __syncthreads();
  float sum = 0.f;
  for (int i = tid; i < L; i += 256) sum += __expf(p[i] - m);
  red[tid] = sum; __syncthreads();
  for (int s = 128; s > 0; s >>= 1) {
    if (tid < s) red[tid] += red[tid + s];
    __syncthreads();
  }
  float inv = 1.f / red[0];
  for (int i = tid; i < L; i += 256) q[i] = __expf(p[i] - m) * inv;
}

// kmeans: gather deterministic initial centroids from RT [B,256,4096]
__global__ void k_km_init(const float* __restrict__ RT, float* __restrict__ cent,
                          long total) {
  long t = (long)blockIdx.x * blockDim.x + threadIdx.x;
  if (t >= total) return;                 // total = B*64*256
  int b = (int)(t / 16384);
  int k = (int)((t / 256) % 64);
  int c = (int)(t % 256);
  int idx = (k * 167 + 29) & 4095;        // 167 coprime with 4096 -> distinct
  cent[t] = RT[(long)b * 1048576 + (long)c * 4096 + idx];
}

// c2[b,k] = sum_c cent^2
__global__ void k_c2(const float* __restrict__ cent, float* __restrict__ c2,
                     long total) {
  long t = (long)blockIdx.x * blockDim.x + threadIdx.x;
  if (t >= total) return;                 // total = B*64
  const float* p = cent + t * 256;
  float s = 0.f;
  for (int c = 0; c < 256; ++c) { float v = p[c]; s += v * v; }
  c2[t] = s;
}

// labels[b,n] = argmin_k (c2[b,k] - 2*dot[b,k,n])
__global__ void k_argmin(const float* __restrict__ dot, const float* __restrict__ c2,
                         int* __restrict__ labels, int N, long total) {
  long t = (long)blockIdx.x * blockDim.x + threadIdx.x;
  if (t >= total) return;                 // total = B*N
  int b = (int)(t / N), n = (int)(t % N);
  const float* dp = dot + (long)b * 64 * N + n;
  const float* cp = c2 + (long)b * 64;
  float best = 1e30f; int bi = 0;
  for (int k = 0; k < 64; ++k) {
    float d = cp[k] - 2.f * dp[(long)k * N];
    if (d < best) { best = d; bi = k; }
  }
  labels[t] = bi;
}

// accumulate cluster sums: cnt[b,k]++, summ[b,k,:] += RT[b,:,n]
__global__ void k_accum(const float* __restrict__ RT, const int* __restrict__ labels,
                        float* __restrict__ cnt, float* __restrict__ summ,
                        int N, long total) {
  long t = (long)blockIdx.x * blockDim.x + threadIdx.x;
  if (t >= total) return;                 // total = B*N
  int b = (int)(t / N), n = (int)(t % N);
  int k = labels[t];
  atomicAdd(&cnt[b * 64 + k], 1.f);
  const float* rt = RT + (long)b * 256 * N + n;
  float* s = summ + ((long)b * 64 + k) * 256;
  for (int c = 0; c < 256; ++c) atomicAdd(&s[c], rt[(long)c * N]);
}

// cent = cnt>0 ? summ/max(cnt,1) : cent
__global__ void k_update(const float* __restrict__ cnt, const float* __restrict__ summ,
                         float* __restrict__ cent, long total) {
  long t = (long)blockIdx.x * blockDim.x + threadIdx.x;
  if (t >= total) return;                 // total = B*64*256
  float c = cnt[t / 256];
  if (c > 0.f) cent[t] = summ[t] / fmaxf(c, 1.f);
}

// refcat[b, m=(r*256+l), c] = ref_protos[r,b,l,c]
__global__ void k_refcat(const float* __restrict__ refP, float* __restrict__ refcat,
                         long total) {
  long t = (long)blockIdx.x * blockDim.x + threadIdx.x;
  if (t >= total) return;                 // total = B*768*256
  int b = (int)(t / 196608);
  int rem = (int)(t % 196608);
  int m = rem / 256, c = rem % 256;
  int r = m / 256, l = m % 256;
  refcat[t] = refP[(((long)r * 8 + b) * 256 + l) * 256 + c];
}

// ---------------------------------------------------------------------------
// Launch orchestration
// ---------------------------------------------------------------------------
static inline dim3 ew(long n) { return dim3((unsigned)((n + 255) / 256)); }

extern "C" void kernel_launch(void* const* d_in, const int* in_sizes, int n_in,
                              void* d_out, int out_size, void* d_ws, size_t ws_size,
                              hipStream_t stream) {
  (void)in_sizes; (void)n_in; (void)out_size; (void)ws_size;
  const int B = 8, CIN = 1024, OC = 256, NS = 16, N = 4096, R = 3, KC = 64;

  const float* target = (const float*)d_in[0];   // [B,1024,64,64]
  const float* refs   = (const float*)d_in[1];   // [R,B,1024,64,64]
  const float* W_red  = (const float*)d_in[2];   // [256,1024]
  const float* b_red  = (const float*)d_in[3];
  const float* W_slot = (const float*)d_in[4];   // [16,1024]
  const float* b_slot = (const float*)d_in[5];
  const float* Wq     = (const float*)d_in[6];
  const float* Wk     = (const float*)d_in[7];
  const float* Wv     = (const float*)d_in[8];
  const float* Wo     = (const float*)d_in[9];

  float* out = (float*)d_out;
  // output slices (flat, reference return order)
  float* o_RT    = out;                                   // [B,256,64,64]
  float* o_AGG   = out + (long)B * OC * N;                // [B,64,256]
  float* o_REFP  = o_AGG + (long)B * KC * OC;             // [R,B,256,256]
  float* o_SLOTP = o_REFP + (long)R * B * OC * OC;        // [B,16,256]
  float* o_SLOTM = o_SLOTP + (long)B * NS * OC;           // [B,16,64,64]

  // workspace bump allocation (floats)
  float* ws = (float*)d_ws;
  long off = 0;
  auto alloc = [&](long n) { float* p = ws + off; off += n; return p; };
  float* w_slotlog = alloc((long)B * NS * N);        // slot logits
  float* w_cent    = alloc((long)B * KC * OC);       // centroids
  float* w_c2      = alloc((long)B * KC);
  float* w_dot     = alloc((long)B * KC * N);        // [B,64,4096]
  float* w_cnt     = alloc((long)B * KC);            // (cnt, summ contiguous)
  float* w_summ    = alloc((long)B * KC * OC);
  int*   w_lab     = (int*)alloc((long)B * N);
  float* w_RR      = alloc((long)B * OC * N);        // reduced ref
  float* w_attnR   = alloc((long)B * OC * N);        // ref softmax
  float* w_refcat  = alloc((long)B * 768 * OC);      // [B,768,256]
  float* w_q       = alloc((long)B * OC * 64);
  float* w_k       = alloc((long)B * OC * 768);
  float* w_v       = alloc((long)B * OC * 768);
  float* w_sc      = alloc((long)B * 64 * 768);
  float* w_agg     = alloc((long)B * OC * 64);

  // ---- 1. reduced_target = W_red x target + b_red  -> o_RT ----
  gemm_wmma_bf16<false,false,false,false,false><<<dim3(64, 8, B), 256, 0, stream>>>(
      W_red, 0, CIN, target, (long)CIN * N, N, o_RT, (long)OC * N, N,
      b_red, OC, N, CIN, 1.f);

  // ---- 2. slot_logits = W_slot x target + b_slot  (M=16 -> guarded) ----
  gemm_wmma_bf16<false,false,false,false,true><<<dim3(64, 1, B), 256, 0, stream>>>(
      W_slot, 0, CIN, target, (long)CIN * N, N, w_slotlog, (long)NS * N, N,
      b_slot, NS, N, CIN, 1.f);

  // ---- 3. slot masks: softmax over 16 slots -> o_SLOTM ----
  k_softmax16<<<ew((long)B * N), 256, 0, stream>>>(w_slotlog, o_SLOTM, N, (long)B * N);

  // ---- 4. slot_protos = masks x RT^T  -> o_SLOTP  (M=16 -> guarded) ----
  gemm_wmma_bf16<false,true,false,false,true><<<dim3(4, 1, B), 256, 0, stream>>>(
      o_SLOTM, (long)NS * N, N, o_RT, (long)OC * N, N, o_SLOTP, (long)NS * OC, OC,
      nullptr, NS, OC, N, 1.f);

  // ---- 5. kmeans on RT points ----
  k_km_init<<<ew((long)B * KC * OC), 256, 0, stream>>>(o_RT, w_cent, (long)B * KC * OC);
  for (int it = 0; it < 10; ++it) {
    k_c2<<<ew((long)B * KC), 256, 0, stream>>>(w_cent, w_c2, (long)B * KC);
    gemm_wmma_bf16<false,false,false,false,false><<<dim3(64, 2, B), 256, 0, stream>>>(
        w_cent, (long)KC * OC, OC, o_RT, (long)OC * N, N, w_dot, (long)KC * N, N,
        nullptr, KC, N, OC, 1.f);
    k_argmin<<<ew((long)B * N), 256, 0, stream>>>(w_dot, w_c2, w_lab, N, (long)B * N);
    k_zero<<<ew((long)B * KC * (OC + 1)), 256, 0, stream>>>(w_cnt, (long)B * KC * (OC + 1));
    k_accum<<<ew((long)B * N), 256, 0, stream>>>(o_RT, w_lab, w_cnt, w_summ, N, (long)B * N);
    k_update<<<ew((long)B * KC * OC), 256, 0, stream>>>(w_cnt, w_summ, w_cent, (long)B * KC * OC);
  }
  // final assignment -> target_protos = cluster sums (one-hot masks x feat)
  k_c2<<<ew((long)B * KC), 256, 0, stream>>>(w_cent, w_c2, (long)B * KC);
  gemm_wmma_bf16<false,false,false,false,false><<<dim3(64, 2, B), 256, 0, stream>>>(
      w_cent, (long)KC * OC, OC, o_RT, (long)OC * N, N, w_dot, (long)KC * N, N,
      nullptr, KC, N, OC, 1.f);
  k_argmin<<<ew((long)B * N), 256, 0, stream>>>(w_dot, w_c2, w_lab, N, (long)B * N);
  k_zero<<<ew((long)B * KC * (OC + 1)), 256, 0, stream>>>(w_cnt, (long)B * KC * (OC + 1));
  k_accum<<<ew((long)B * N), 256, 0, stream>>>(o_RT, w_lab, w_cnt, w_summ, N, (long)B * N);

  // residual copy: aggregated = target_protos ([B,64,256]), gemm adds on top
  k_copy<<<ew((long)B * KC * OC), 256, 0, stream>>>(w_summ, o_AGG, (long)B * KC * OC);

  // ---- 6. per-ref: reduce, row-softmax over N, protos -> o_REFP ----
  for (int r = 0; r < R; ++r) {
    const float* refb = refs + (long)r * B * CIN * N;
    gemm_wmma_bf16<false,false,false,false,false><<<dim3(64, 8, B), 256, 0, stream>>>(
        W_red, 0, CIN, refb, (long)CIN * N, N, w_RR, (long)OC * N, N,
        b_red, OC, N, CIN, 1.f);
    k_row_softmax<<<dim3((unsigned)(B * OC)), 256, 0, stream>>>(w_RR, w_attnR, N);
    gemm_wmma_bf16<false,true,false,false,false><<<dim3(4, 8, B), 256, 0, stream>>>(
        w_attnR, (long)OC * N, N, w_RR, (long)OC * N, N,
        o_REFP + (long)r * B * OC * OC, (long)OC * OC, OC,
        nullptr, OC, OC, N, 1.f);
  }

  // ---- 7. concat refs: refcat [B, M=768, C=256] ----
  k_refcat<<<ew((long)B * 768 * OC), 256, 0, stream>>>(o_REFP, w_refcat, (long)B * 768 * OC);

  // ---- 8. feature aggregator ----
  // q = Wq x tp^T  ([B,256,64])
  gemm_wmma_bf16<false,true,false,false,false><<<dim3(1, 8, B), 256, 0, stream>>>(
      Wq, 0, OC, w_summ, (long)KC * OC, OC, w_q, (long)OC * 64, 64,
      nullptr, OC, 64, OC, 1.f);
  // k = Wk x refcat^T  ([B,256,768])
  gemm_wmma_bf16<false,true,false,false,false><<<dim3(12, 8, B), 256, 0, stream>>>(
      Wk, 0, OC, w_refcat, (long)768 * OC, OC, w_k, (long)OC * 768, 768,
      nullptr, OC, 768, OC, 1.f);
  // v = Wv x refcat^T
  gemm_wmma_bf16<false,true,false,false,false><<<dim3(12, 8, B), 256, 0, stream>>>(
      Wv, 0, OC, w_refcat, (long)768 * OC, OC, w_v, (long)OC * 768, 768,
      nullptr, OC, 768, OC, 1.f);
  // scores = q^T x k / 16  ([B,64,768])
  gemm_wmma_bf16<true,false,false,false,false><<<dim3(12, 2, B), 256, 0, stream>>>(
      w_q, (long)OC * 64, 64, w_k, (long)OC * 768, 768, w_sc, (long)64 * 768, 768,
      nullptr, 64, 768, OC, 1.f / 16.f);
  // softmax over M=768 (in place)
  k_row_softmax<<<dim3((unsigned)(B * 64)), 256, 0, stream>>>(w_sc, w_sc, 768);
  // agg = v x attn^T  ([B,256,64])
  gemm_wmma_bf16<false,true,false,false,false><<<dim3(1, 8, B), 256, 0, stream>>>(
      w_v, (long)OC * 768, 768, w_sc, (long)64 * 768, 768, w_agg, (long)OC * 64, 64,
      nullptr, OC, 64, 768, 1.f);
  // aggregated += (Wo x agg)^T  (store transposed into [B,64,256])
  gemm_wmma_bf16<false,false,true,true,false><<<dim3(1, 8, B), 256, 0, stream>>>(
      Wo, 0, OC, w_agg, (long)OC * 64, 64, o_AGG, (long)KC * OC, OC,
      nullptr, OC, 64, OC, 1.f);
}
